// Loss_15324443312557
// MI455X (gfx1250) — matrix operand order repaired and checked by reading, not compiled, not run on hardware
//
#include <hip/hip_runtime.h>
#include <hip/hip_bf16.h>

#define B_ROWS 8192
#define DIM    2048
#define INV_T  (1.0f/0.07f)
#define TEMP_  0.07f

#define BM 128
#define BN 256
#define BK 64
#define LDK 72            // padded LDS row stride (ushorts): 144B, 16B-aligned
#define NCHUNK 16
#define CHUNK_COLS 512

#define AS1 __attribute__((address_space(1)))
#define AS3 __attribute__((address_space(3)))

#if __has_builtin(__builtin_amdgcn_global_load_async_to_lds_b128)
#define HAVE_ASYNC_LDS 1
#else
#define HAVE_ASYNC_LDS 0
#endif

typedef __attribute__((ext_vector_type(16))) __bf16 v16bf;
typedef __attribute__((ext_vector_type(8)))  float  v8f;
typedef int gv4i __attribute__((vector_size(16)));   // matches builtin's V4i param type

static __device__ __forceinline__ float bf16_to_f32(unsigned int h) {
    return __uint_as_float(h << 16);
}
static __device__ __forceinline__ unsigned short f32_to_bf16(float f) {
    unsigned int u = __float_as_uint(f);
    u += 0x7FFFu + ((u >> 16) & 1u);  // round to nearest even
    return (unsigned short)(u >> 16);
}

static __device__ __forceinline__ void wait_async_all() {
#if HAVE_ASYNC_LDS
#if __has_builtin(__builtin_amdgcn_s_wait_asynccnt)
    __builtin_amdgcn_s_wait_asynccnt(0);
#else
    asm volatile("s_wait_asynccnt 0x0" ::: "memory");
#endif
#endif
}

// ---------------- kernel 1: f32 -> bf16 convert (8 elems/thread) ----------------
__global__ void k_convert_bf16(const float* __restrict__ src, unsigned short* __restrict__ dst) {
    size_t i = ((size_t)blockIdx.x * blockDim.x + threadIdx.x) * 8;
    float4 a = *(const float4*)(src + i);
    float4 b = *(const float4*)(src + i + 4);
    uint4 o;
    o.x = (unsigned int)f32_to_bf16(a.x) | ((unsigned int)f32_to_bf16(a.y) << 16);
    o.y = (unsigned int)f32_to_bf16(a.z) | ((unsigned int)f32_to_bf16(a.w) << 16);
    o.z = (unsigned int)f32_to_bf16(b.x) | ((unsigned int)f32_to_bf16(b.y) << 16);
    o.w = (unsigned int)f32_to_bf16(b.z) | ((unsigned int)f32_to_bf16(b.w) << 16);
    *(uint4*)(dst + i) = o;
}

// Stage one K-phase of A (128x64) and B (256x64) bf16 tiles into LDS.
// Async path: per-lane GLOBAL_LOAD_ASYNC_TO_LDS_B128; the shared IOFFSET (literal
// immediate) advances both global and LDS addresses, so one base pair suffices.
static __device__ __forceinline__ void stage_tiles(const unsigned short* __restrict__ xb,
                                                   int rowbase, int colbase, int k0,
                                                   unsigned short* as, unsigned short* bs,
                                                   int tid) {
    const int r   = tid >> 1;
    const int seg = (tid & 1) * 32;
    const unsigned short* ga = xb + (size_t)(rowbase + r) * DIM + k0 + seg;
    const unsigned short* gb = xb + (size_t)(colbase + tid) * DIM + k0;
    unsigned short* la = &as[r * LDK + seg];
    unsigned short* lb = &bs[tid * LDK];
#if HAVE_ASYNC_LDS
    // A tile: 32 bf16 per thread = 4x b128, immediate offsets 0..48
    __builtin_amdgcn_global_load_async_to_lds_b128((AS1 gv4i*)ga, (AS3 gv4i*)la,  0, 0);
    __builtin_amdgcn_global_load_async_to_lds_b128((AS1 gv4i*)ga, (AS3 gv4i*)la, 16, 0);
    __builtin_amdgcn_global_load_async_to_lds_b128((AS1 gv4i*)ga, (AS3 gv4i*)la, 32, 0);
    __builtin_amdgcn_global_load_async_to_lds_b128((AS1 gv4i*)ga, (AS3 gv4i*)la, 48, 0);
    // B tile: 64 bf16 per thread = 8x b128, immediate offsets 0..112
    __builtin_amdgcn_global_load_async_to_lds_b128((AS1 gv4i*)gb, (AS3 gv4i*)lb,   0, 0);
    __builtin_amdgcn_global_load_async_to_lds_b128((AS1 gv4i*)gb, (AS3 gv4i*)lb,  16, 0);
    __builtin_amdgcn_global_load_async_to_lds_b128((AS1 gv4i*)gb, (AS3 gv4i*)lb,  32, 0);
    __builtin_amdgcn_global_load_async_to_lds_b128((AS1 gv4i*)gb, (AS3 gv4i*)lb,  48, 0);
    __builtin_amdgcn_global_load_async_to_lds_b128((AS1 gv4i*)gb, (AS3 gv4i*)lb,  64, 0);
    __builtin_amdgcn_global_load_async_to_lds_b128((AS1 gv4i*)gb, (AS3 gv4i*)lb,  80, 0);
    __builtin_amdgcn_global_load_async_to_lds_b128((AS1 gv4i*)gb, (AS3 gv4i*)lb,  96, 0);
    __builtin_amdgcn_global_load_async_to_lds_b128((AS1 gv4i*)gb, (AS3 gv4i*)lb, 112, 0);
#else
    #pragma unroll
    for (int i = 0; i < 4; ++i)
        *(uint4*)(la + i * 8) = *(const uint4*)(ga + i * 8);
    #pragma unroll
    for (int i = 0; i < 8; ++i)
        *(uint4*)(lb + i * 8) = *(const uint4*)(gb + i * 8);
#endif
}

// ---------------- kernel 2: fused Gram-GEMM + online row max / sum-exp ----------------
// grid = (B/BM, NCHUNK). Each WG: 128 rows x 512 cols (two 256-col tiles), K over 2048.
// Double-buffered LDS, async global->LDS copies overlapped with WMMA issue.
__global__ __launch_bounds__(256)
void k_rowstats(const unsigned short* __restrict__ xb,
                float* __restrict__ pmax, float* __restrict__ psum) {
    __shared__ unsigned short As[2][BM * LDK];
    __shared__ unsigned short Bs[2][BN * LDK];
    __shared__ float sM[BM * 4];
    __shared__ float sS[BM * 4];

    const int tid  = threadIdx.x;
    const int lane = tid & 31;
    const int wave = tid >> 5;
    const int wm   = wave >> 2;   // 0..1 : row half of the 128-row tile
    const int wn   = wave & 3;    // 0..3 : 64-col slice of the 256-col tile
    const int half = lane >> 4;
    const int ln   = lane & 15;

    const int rowbase = blockIdx.x * BM;
    const int chunk   = blockIdx.y;

    for (int i = tid; i < BM * 4; i += 256) { sM[i] = -__builtin_inff(); sS[i] = 0.f; }
    __syncthreads();

    for (int ct = 0; ct < 2; ++ct) {
        const int colbase = chunk * CHUNK_COLS + ct * BN;

        v8f acc[4][4];
        #pragma unroll
        for (int i = 0; i < 4; ++i)
            #pragma unroll
            for (int j = 0; j < 4; ++j)
                acc[i][j] = (v8f){0.f,0.f,0.f,0.f,0.f,0.f,0.f,0.f};

        const int NT = DIM / BK;  // 32 K-phases
        stage_tiles(xb, rowbase, colbase, 0, As[0], Bs[0], tid);

        for (int t = 0; t < NT; ++t) {
            const int buf = t & 1;
            wait_async_all();     // my copies into As/Bs[buf] landed
            __syncthreads();      // everyone's copies landed; prior readers done
            if (t + 1 < NT)       // prefetch next phase into the other buffer
                stage_tiles(xb, rowbase, colbase, (t + 1) * BK, As[buf ^ 1], Bs[buf ^ 1], tid);

            const unsigned short* as = As[buf];
            const unsigned short* bs = Bs[buf];
            #pragma unroll
            for (int kk = 0; kk < BK; kk += 32) {
                // A fragments: 16x32 bf16; lane half selects K sub-octets per ISA layout
                union { uint4 q[2]; v16bf v; } af[4];
                #pragma unroll
                for (int tM = 0; tM < 4; ++tM) {
                    const int m = wm * 64 + tM * 16 + ln;
                    af[tM].q[0] = *(const uint4*)&as[m * LDK + kk + half * 8];
                    af[tM].q[1] = *(const uint4*)&as[m * LDK + kk + 16 + half * 8];
                }
                #pragma unroll
                for (int tN = 0; tN < 4; ++tN) {
                    // B fragment: column n = out-row (Gram), same K sub-octets as A
                    union { uint4 q[2]; v16bf v; } bfr;
                    const int n = wn * 64 + tN * 16 + ln;
                    bfr.q[0] = *(const uint4*)&bs[n * LDK + kk + half * 8];
                    bfr.q[1] = *(const uint4*)&bs[n * LDK + kk + 16 + half * 8];
                    #pragma unroll
                    for (int tM = 0; tM < 4; ++tM) {
                        acc[tM][tN] = __builtin_amdgcn_wmma_f32_16x16x32_bf16(
                            false, af[tM].v, false, bfr.v, (short)0, acc[tM][tN],
                            false, false);
                    }
                }
            }
        }

        // flash-style tile stats: per C/D layout, VGPR v holds row m = 16*tM + v + 8*half,
        // its 16 column values live across the lanes of one half-wave.
        #pragma unroll
        for (int tM = 0; tM < 4; ++tM) {
            #pragma unroll
            for (int v = 0; v < 8; ++v) {
                float x0 = acc[tM][0][v] * INV_T;
                float x1 = acc[tM][1][v] * INV_T;
                float x2 = acc[tM][2][v] * INV_T;
                float x3 = acc[tM][3][v] * INV_T;
                float mx = fmaxf(fmaxf(x0, x1), fmaxf(x2, x3));
                #pragma unroll
                for (int off = 1; off <= 8; off <<= 1)
                    mx = fmaxf(mx, __shfl_xor(mx, off, 32));
                float s = __expf(x0 - mx) + __expf(x1 - mx) + __expf(x2 - mx) + __expf(x3 - mx);
                #pragma unroll
                for (int off = 1; off <= 8; off <<= 1)
                    s += __shfl_xor(s, off, 32);
                if (ln == 0) {  // one writer per (row, wn): no races, deterministic
                    const int m = wm * 64 + tM * 16 + v + half * 8;
                    float oM = sM[m * 4 + wn], oS = sS[m * 4 + wn];
                    float nM = fmaxf(oM, mx);
                    sS[m * 4 + wn] = oS * __expf(oM - nM) + s * __expf(mx - nM);
                    sM[m * 4 + wn] = nM;
                }
            }
        }
    }
    __syncthreads();
    if (tid < BM) {  // deterministic merge of the 4 wn slices -> per-chunk partials
        float M = -__builtin_inff(), S = 0.f;
        #pragma unroll
        for (int c = 0; c < 4; ++c) {
            float m = sM[tid * 4 + c], s = sS[tid * 4 + c];
            float nM = fmaxf(M, m);
            S = S * __expf(M - nM) + s * __expf(m - nM);
            M = nM;
        }
        pmax[(size_t)chunk * B_ROWS + rowbase + tid] = M;
        psum[(size_t)chunk * B_ROWS + rowbase + tid] = S;
    }
}

// ---------------- kernel 3: per-row self-dot (diagonal term), one wave per row ----------------
__global__ void k_selfdot(const unsigned short* __restrict__ xb, float* __restrict__ selfdot) {
    const int wave = threadIdx.x >> 5;
    const int lane = threadIdx.x & 31;
    const int r = blockIdx.x * 8 + wave;
    const unsigned short* g = xb + (size_t)r * DIM;
    float accu = 0.f;
    #pragma unroll
    for (int i = 0; i < 8; ++i) {
        uint4 q = *(const uint4*)(g + (i * 32 + lane) * 8);
        unsigned int u[4] = {q.x, q.y, q.z, q.w};
        #pragma unroll
        for (int j = 0; j < 4; ++j) {
            float lo = bf16_to_f32(u[j] & 0xFFFFu);
            float hi = bf16_to_f32(u[j] >> 16);
            accu = fmaf(lo, lo, accu);
            accu = fmaf(hi, hi, accu);
        }
    }
    #pragma unroll
    for (int off = 1; off <= 16; off <<= 1) accu += __shfl_xor(accu, off, 32);
    if (lane == 0) selfdot[r] = accu * INV_T;
}

// ---------------- kernel 4: merge chunk partials, remove diagonal, store M and log S ----------------
__global__ void k_merge(const float* __restrict__ pmax, const float* __restrict__ psum,
                        const float* __restrict__ selfdot,
                        float* __restrict__ rowM, float* __restrict__ rowLS) {
    int r = blockIdx.x * blockDim.x + threadIdx.x;
    if (r >= B_ROWS) return;
    float M = -__builtin_inff();
    for (int c = 0; c < NCHUNK; ++c) M = fmaxf(M, pmax[(size_t)c * B_ROWS + r]);
    float S = 0.f;
    for (int c = 0; c < NCHUNK; ++c)
        S += psum[(size_t)c * B_ROWS + r] * __expf(pmax[(size_t)c * B_ROWS + r] - M);
    S -= __expf(selfdot[r] - M);          // diagonal excluded from denominator only
    S = fmaxf(S, 1e-30f);
    rowM[r] = M;
    rowLS[r] = __logf(S);
}

// ---------------- kernel 5a: zero scratch accumulators ----------------
__global__ void k_zero(float* __restrict__ p, int n) {
    int i = blockIdx.x * blockDim.x + threadIdx.x;
    if (i < n) p[i] = 0.f;
}

// ---------------- kernel 5b: sparse pair log-probs + segment scatter-add ----------------
__global__ void k_pairs(const unsigned short* __restrict__ xb,
                        const int* __restrict__ row, const int* __restrict__ col,
                        const float* __restrict__ rowM, const float* __restrict__ rowLS,
                        float* __restrict__ lpSum, float* __restrict__ lpCnt, int nnz) {
    const int wave = threadIdx.x >> 5;
    const int lane = threadIdx.x & 31;
    const int p = blockIdx.x * 8 + wave;
    if (p >= nnz) return;
    const int r = row[p], c = col[p];
    const unsigned short* ga = xb + (size_t)r * DIM;
    const unsigned short* gb = xb + (size_t)c * DIM;
    float accu = 0.f;
    #pragma unroll
    for (int i = 0; i < 8; ++i) {
        const int off = (i * 32 + lane) * 8;
        uint4 qa = *(const uint4*)(ga + off);
        uint4 qb = *(const uint4*)(gb + off);
        unsigned int ua[4] = {qa.x, qa.y, qa.z, qa.w};
        unsigned int ub[4] = {qb.x, qb.y, qb.z, qb.w};
        #pragma unroll
        for (int j = 0; j < 4; ++j) {
            accu = fmaf(bf16_to_f32(ua[j] & 0xFFFFu), bf16_to_f32(ub[j] & 0xFFFFu), accu);
            accu = fmaf(bf16_to_f32(ua[j] >> 16),     bf16_to_f32(ub[j] >> 16),     accu);
        }
    }
    #pragma unroll
    for (int off = 1; off <= 16; off <<= 1) accu += __shfl_xor(accu, off, 32);
    if (lane == 0) {
        float lp = accu * INV_T - rowM[r] - rowLS[r];
        atomicAdd(&lpSum[r], lp);
        atomicAdd(&lpCnt[r], 1.0f);
    }
}

// ---------------- kernel 6: final scalar reduction ----------------
__global__ void k_finish(const float* __restrict__ lpSum, const float* __restrict__ lpCnt,
                         float* __restrict__ out) {
    __shared__ float sl[256], sp[256];
    float lsum = 0.f, pres = 0.f;
    for (int i = threadIdx.x; i < B_ROWS; i += 256) {
        float c = lpCnt[i];
        if (c > 0.f) { lsum += -(lpSum[i] / c) * TEMP_; pres += 1.f; }
    }
    sl[threadIdx.x] = lsum; sp[threadIdx.x] = pres;
    __syncthreads();
    for (int s = 128; s > 0; s >>= 1) {
        if (threadIdx.x < s) { sl[threadIdx.x] += sl[threadIdx.x + s]; sp[threadIdx.x] += sp[threadIdx.x + s]; }
        __syncthreads();
    }
    if (threadIdx.x == 0) out[0] = (sp[0] > 0.f) ? (sl[0] / sp[0]) : 0.f;
}

extern "C" void kernel_launch(void* const* d_in, const int* in_sizes, int n_in,
                              void* d_out, int out_size, void* d_ws, size_t ws_size,
                              hipStream_t stream) {
    (void)n_in; (void)out_size; (void)ws_size;
    const float* x   = (const float*)d_in[0];
    const int*   row = (const int*)d_in[1];
    const int*   col = (const int*)d_in[2];
    const int    nnz = in_sizes[1];
    float* out = (float*)d_out;

    // workspace carve-up (~34.7 MB total)
    char* ws = (char*)d_ws;
    unsigned short* xb = (unsigned short*)ws;  ws += (size_t)B_ROWS * DIM * sizeof(unsigned short);
    float* pmax    = (float*)ws; ws += (size_t)NCHUNK * B_ROWS * sizeof(float);
    float* psum    = (float*)ws; ws += (size_t)NCHUNK * B_ROWS * sizeof(float);
    float* selfdot = (float*)ws; ws += B_ROWS * sizeof(float);
    float* rowM    = (float*)ws; ws += B_ROWS * sizeof(float);
    float* rowLS   = (float*)ws; ws += B_ROWS * sizeof(float);
    float* lpSum   = (float*)ws; ws += B_ROWS * sizeof(float);
    float* lpCnt   = (float*)ws; ws += B_ROWS * sizeof(float);

    k_convert_bf16<<<(B_ROWS * DIM) / (256 * 8), 256, 0, stream>>>(x, xb);
    k_rowstats<<<dim3(B_ROWS / BM, NCHUNK), 256, 0, stream>>>(xb, pmax, psum);
    k_selfdot<<<B_ROWS / 8, 256, 0, stream>>>(xb, selfdot);
    k_merge<<<(B_ROWS + 255) / 256, 256, 0, stream>>>(pmax, psum, selfdot, rowM, rowLS);
    k_zero<<<(2 * B_ROWS + 255) / 256, 256, 0, stream>>>(lpSum, 2 * B_ROWS); // lpSum+lpCnt contiguous
    k_pairs<<<(nnz + 7) / 8, 256, 0, stream>>>(xb, row, col, rowM, rowLS, lpSum, lpCnt, nnz);
    k_finish<<<1, 256, 0, stream>>>(lpSum, lpCnt, out);
}
